// AddressNER_23897198035408
// MI455X (gfx1250) — compile-verified, matrix operand order checked by reading, not run on hardware
//
#include <hip/hip_runtime.h>
#include <hip/hip_bf16.h>
#include <math.h>

// ---- model dims -----------------------------------------------------------
#define B_   256
#define S_   128
#define H_   768
#define NH_  12
#define DH_  64
#define F_   3072
#define L_   2
#define P_   100
#define LH_  256
#define NL_  57
#define BS_  (B_*S_)      // 32768
#define D1P_ 896          // H+P padded to mult of 32
#define G4_  1024         // 4*LH

typedef _Float16 h16;
typedef __attribute__((ext_vector_type(16))) _Float16 v16h;
typedef __attribute__((ext_vector_type(8)))  _Float16 v8h;
typedef __attribute__((ext_vector_type(8)))  float    v8f;
typedef __attribute__((ext_vector_type(4))) unsigned int u32x4;
typedef __attribute__((ext_vector_type(8))) int          i32x8;
typedef __attribute__((ext_vector_type(4))) int          i32x4;
typedef __attribute__((address_space(3))) void lds_void;

// ---- CDNA5 async global->LDS copy (ASYNCcnt) ------------------------------
__device__ __forceinline__ void async_b128(void* lds, const void* g) {
    lds_void* lp = (lds_void*)lds;
    asm volatile("global_load_async_to_lds_b128 %0, %1, off"
                 :: "v"(lp), "v"(g) : "memory");
}
__device__ __forceinline__ void wait_async_le2() {
    asm volatile("s_wait_asynccnt 0x2" ::: "memory");
}
__device__ __forceinline__ void wait_async_le0() {
    asm volatile("s_wait_asynccnt 0x0" ::: "memory");
}

// ---- CDNA5 Tensor Data Mover: 2D tile global->LDS with hardware padding ---
#if __has_builtin(__builtin_amdgcn_tensor_load_to_lds)
#define HAVE_TDM 1
__device__ __forceinline__ void tdm_load_2d(void* lds_dst, const void* gsrc,
                                            unsigned rows, unsigned rowlen,
                                            long row_stride,
                                            int pad_interval_code, int pad_amount_code)
{
    unsigned lds = (unsigned)(size_t)(lds_void*)lds_dst;
    unsigned long long ga = (unsigned long long)(size_t)gsrc;
    u32x4 g0;
    g0[0] = 1u;                                        // count=1, user mode
    g0[1] = lds;                                       // lds_addr
    g0[2] = (unsigned)(ga & 0xffffffffull);            // global_addr[31:0]
    g0[3] = (unsigned)((ga >> 32) & 0x1ffffffull) | (2u << 30);  // ga[56:32] | type=2
    i32x8 g1;
    unsigned w0 = (1u << 16);                          // data_size = 2 bytes
    if (pad_amount_code >= 0)
        w0 |= (1u << 20) | ((unsigned)pad_interval_code << 22)
            | ((unsigned)pad_amount_code << 25);       // pad_enable/interval/amount
    g1[0] = (int)w0;
    g1[1] = (int)((rowlen & 0xffffu) << 16);           // tensor_dim0[15:0]
    g1[2] = (int)(((rowlen >> 16) & 0xffffu) | ((rows & 0xffffu) << 16)); // dim0 hi | dim1 lo
    g1[3] = (int)(((rows >> 16) & 0xffffu) | ((rowlen & 0xffffu) << 16)); // dim1 hi | tile_dim0
    g1[4] = (int)(rows & 0xffffu);                     // tile_dim1 | tile_dim2=0
    g1[5] = (int)(unsigned)(row_stride & 0xffffffffll);          // stride0[31:0]
    g1[6] = (int)(unsigned)((row_stride >> 32) & 0xffffll);      // stride0[47:32] | stride1 lo = 0
    g1[7] = 0;
    i32x4 z4 = {0, 0, 0, 0};
#if __has_include(<hip/amd_detail/amd_gfx1250_TDM.h>)
    i32x8 z8 = {0, 0, 0, 0, 0, 0, 0, 0};
    __builtin_amdgcn_tensor_load_to_lds(g0, g1, z4, z4, z8, 0);
#else
    __builtin_amdgcn_tensor_load_to_lds(g0, g1, z4, z4, 0);
#endif
}
#endif

// ---- WMMA helpers ---------------------------------------------------------
// A-fragment (16x32 f16, wave32): lane = m + 16*half; element j holds
// K = half*8 + j (j<8) or 16 + half*8 + (j-8). Two 16B contiguous loads.
__device__ __forceinline__ v16h frag_ld(const h16* base, int ld) {
    int lane = threadIdx.x & 31;
    int m  = lane & 15;
    int hs = lane >> 4;
    const h16* p = base + (size_t)m * ld + hs * 8;
    v8h lo = *(const v8h*)p;
    v8h hi = *(const v8h*)(p + 16);
    v16h r;
#pragma unroll
    for (int j = 0; j < 8; ++j) { r[j] = lo[j]; r[j + 8] = hi[j]; }
    return r;
}

__device__ __forceinline__ v8f wmma16(v16h a, v16h b, v8f c) {
    return __builtin_amdgcn_wmma_f32_16x16x32_f16(false, a, false, b,
                                                  (short)0, c, false, false);
}

__device__ __forceinline__ float gelu_f(float x) {
    return 0.5f * x * (1.0f + tanhf(0.7978845608f * (x + 0.044715f * x * x * x)));
}
__device__ __forceinline__ float sigm_f(float x) { return 1.0f / (1.0f + expf(-x)); }

// ---- generic f16 WMMA GEMM: C = A(MxK) * BT(NxK) (+bias)(+gelu) -----------
// block = 256 threads = 8 waves; 64x64 block tile; wave tile 16x32.
// Double-buffered async global->LDS tile pipeline (ASYNCcnt).
__launch_bounds__(256)
__global__ void gemm_kernel(const h16* __restrict__ A, const h16* __restrict__ BT,
                            const float* __restrict__ bias,
                            float* __restrict__ Cf, h16* __restrict__ Ch,
                            int M, int N, int K, int do_gelu)
{
    __shared__ alignas(16) h16 As[2][64 * 40];
    __shared__ alignas(16) h16 Bs[2][64 * 40];
    int tid = threadIdx.x;
    int n0 = blockIdx.x * 64;
    int m0 = blockIdx.y * 64;
    int wid = tid >> 5, lane = tid & 31;
    int wm = wid & 3, wn = wid >> 2;

    int r = tid >> 2, cg = (tid & 3) * 8;          // 64 rows x 4 x 8-half chunks
    const h16* gA = A + (size_t)(m0 + r) * K + cg;
    const h16* gB = BT + (size_t)(n0 + r) * K + cg;
    int loff = r * 40 + cg;

    async_b128(&As[0][loff], gA);
    async_b128(&Bs[0][loff], gB);

    v8f acc0 = {}; v8f acc1 = {};
    for (int k0 = 0; k0 < K; k0 += 32) {
        int cur = (k0 >> 5) & 1;
        int nxt = cur ^ 1;
        if (k0 + 32 < K) {
            async_b128(&As[nxt][loff], gA + k0 + 32);
            async_b128(&Bs[nxt][loff], gB + k0 + 32);
            wait_async_le2();   // the 2 just-issued may remain in flight
        } else {
            wait_async_le0();
        }
        __syncthreads();
        v16h a  = frag_ld(&As[cur][wm * 16 * 40], 40);
        v16h b0 = frag_ld(&Bs[cur][(wn * 32) * 40], 40);
        v16h b1 = frag_ld(&Bs[cur][(wn * 32 + 16) * 40], 40);
        acc0 = wmma16(a, b0, acc0);
        acc1 = wmma16(a, b1, acc1);
        __syncthreads();
    }
    int rbase = m0 + wm * 16 + ((lane >> 4) << 3);
    int cb = n0 + wn * 32 + (lane & 15);
#pragma unroll
    for (int rr = 0; rr < 8; ++rr) {
        int row = rbase + rr;
        float v0 = acc0[rr], v1 = acc1[rr];
        int c0 = cb, c1 = cb + 16;
        if (bias) { v0 += bias[c0]; v1 += bias[c1]; }
        if (do_gelu) { v0 = gelu_f(v0); v1 = gelu_f(v1); }
        if (Cf) { Cf[(size_t)row * N + c0] = v0; Cf[(size_t)row * N + c1] = v1; }
        if (Ch) { Ch[(size_t)row * N + c0] = (h16)v0; Ch[(size_t)row * N + c1] = (h16)v1; }
    }
}

// ---- attention: per (b, head); Q/K staged by TDM, two WMMA GEMMs ----------
__launch_bounds__(256)
__global__ void attn_kernel(const h16* __restrict__ qkv, const int* __restrict__ mask,
                            h16* __restrict__ ctx)
{
    extern __shared__ char smem[];
    h16*   Qs = (h16*)smem;                // 128 x 72
    h16*   Ks = Qs + 128 * 72;             // 128 x 72
    h16*   Vt = Ks + 128 * 72;             // 64 x 136  (V transposed)
    float* Sf = (float*)(Vt + 64 * 136);   // 128 x 132 (f32 scores)
    h16*   Ph = (h16*)(Sf + 128 * 132);    // 128 x 136 (f16 probs)
    float* kb = (float*)(Ph + 128 * 136);  // 128 key-bias

    int tid = threadIdx.x;
    int b = blockIdx.x / NH_;
    int hh = blockIdx.x % NH_;
    const h16* base = qkv + (size_t)(b * S_) * (3 * H_) + hh * DH_;
#ifdef HAVE_TDM
    if (tid < 32) {   // wave 0 drives the Tensor Data Mover (EXEC ignored)
        // pad_interval code 4 = 32 DWORDs (64 halfs), pad_amount code 3 = 4 DWORDs
        // -> LDS row stride 72 halfs, matching the fragment loader.
        tdm_load_2d(Qs, base,      S_, DH_, 3 * H_, 4, 3);
        tdm_load_2d(Ks, base + H_, S_, DH_, 3 * H_, 4, 3);
        __builtin_amdgcn_s_wait_tensorcnt(0);
    }
    {
        int r = tid >> 1, cg = (tid & 1) * 32;
        const h16* vrow = base + (size_t)r * (3 * H_) + 2 * H_;
        for (int j = 0; j < 32; ++j) Vt[(cg + j) * 136 + r] = vrow[cg + j];
    }
#else
    {
        int r = tid >> 1, cg = (tid & 1) * 32;
        const h16* qrow = base + (size_t)r * (3 * H_);
#pragma unroll
        for (int j = 0; j < 32; j += 8) {
            *(v8h*)(Qs + r * 72 + cg + j) = *(const v8h*)(qrow + cg + j);
            *(v8h*)(Ks + r * 72 + cg + j) = *(const v8h*)(qrow + H_ + cg + j);
        }
        const h16* vrow = qrow + 2 * H_;
        for (int j = 0; j < 32; ++j) Vt[(cg + j) * 136 + r] = vrow[cg + j];
    }
#endif
    if (tid < S_) kb[tid] = mask[b * S_ + tid] ? 0.0f : -1e9f;
    __syncthreads();

    int wid = tid >> 5, lane = tid & 31;
    int q0 = wid * 16;
    {   // S = Q K^T * scale + bias
        v8f acc[8] = {};
        for (int kc = 0; kc < DH_; kc += 32) {
            v16h a = frag_ld(Qs + q0 * 72 + kc, 72);
#pragma unroll
            for (int nt = 0; nt < 8; ++nt) {
                v16h bf = frag_ld(Ks + nt * 16 * 72 + kc, 72);
                acc[nt] = wmma16(a, bf, acc[nt]);
            }
        }
        int rb = q0 + ((lane >> 4) << 3);
        int cc = lane & 15;
#pragma unroll
        for (int nt = 0; nt < 8; ++nt) {
            int col = nt * 16 + cc;
#pragma unroll
            for (int rr = 0; rr < 8; ++rr)
                Sf[(rb + rr) * 132 + col] = acc[nt][rr] * 0.125f + kb[col];
        }
    }
    __syncthreads();
    if (tid < S_) {   // row softmax (f32)
        float mx = -1e30f;
        for (int j = 0; j < S_; ++j) mx = fmaxf(mx, Sf[tid * 132 + j]);
        float sum = 0.f;
        for (int j = 0; j < S_; ++j) sum += expf(Sf[tid * 132 + j] - mx);
        float inv = 1.0f / sum;
        for (int j = 0; j < S_; ++j)
            Ph[tid * 136 + j] = (h16)(expf(Sf[tid * 132 + j] - mx) * inv);
    }
    __syncthreads();
    {   // ctx = P @ V
        v8f acc[4] = {};
        for (int kc = 0; kc < S_; kc += 32) {
            v16h a = frag_ld(Ph + q0 * 136 + kc, 136);
#pragma unroll
            for (int nt = 0; nt < 4; ++nt) {
                v16h bf = frag_ld(Vt + nt * 16 * 136 + kc, 136);
                acc[nt] = wmma16(a, bf, acc[nt]);
            }
        }
        int rb = q0 + ((lane >> 4) << 3);
        int cc = lane & 15;
#pragma unroll
        for (int nt = 0; nt < 4; ++nt)
#pragma unroll
            for (int rr = 0; rr < 8; ++rr)
                ctx[(size_t)(b * S_ + rb + rr) * H_ + hh * DH_ + nt * 16 + cc] = (h16)acc[nt][rr];
    }
}

// ---- LayerNorm (H=768), optional residual, dual f32/f16 output ------------
__launch_bounds__(256)
__global__ void ln_kernel(const float* __restrict__ in, const float* __restrict__ res,
                          const float* __restrict__ g, const float* __restrict__ bb,
                          float* __restrict__ of, h16* __restrict__ oh)
{
    __shared__ float s1[256], s2[256];
    int row = blockIdx.x, tid = threadIdx.x;
    float x[3];
    float a = 0.f, q = 0.f;
#pragma unroll
    for (int e = 0; e < 3; ++e) {
        int i = tid + e * 256;
        float v = in[(size_t)row * H_ + i];
        if (res) v += res[(size_t)row * H_ + i];
        x[e] = v; a += v; q += v * v;
    }
    s1[tid] = a; s2[tid] = q; __syncthreads();
    for (int s = 128; s > 0; s >>= 1) {
        if (tid < s) { s1[tid] += s1[tid + s]; s2[tid] += s2[tid + s]; }
        __syncthreads();
    }
    float mean = s1[0] / (float)H_;
    float var = s2[0] / (float)H_ - mean * mean;
    float rstd = rsqrtf(var + 1e-12f);
#pragma unroll
    for (int e = 0; e < 3; ++e) {
        int i = tid + e * 256;
        float y = (x[e] - mean) * rstd * g[i] + bb[i];
        if (of) of[(size_t)row * H_ + i] = y;
        if (oh) oh[(size_t)row * H_ + i] = (h16)y;
    }
}

// ---- embedding ------------------------------------------------------------
__global__ void embed_kernel(const int* __restrict__ ids, const float* __restrict__ we,
                             const float* __restrict__ pos, float* __restrict__ out)
{
    int row = blockIdx.x, tid = threadIdx.x;
    int id = ids[row];
    int s = row % S_;
#pragma unroll
    for (int e = 0; e < 3; ++e) {
        int i = tid + e * 256;
        out[(size_t)row * H_ + i] = we[(size_t)id * H_ + i] + pos[(size_t)s * H_ + i];
    }
}

// ---- weight packing: f32 (K x N) -> f16 transposed (N x Kp), K zero-padded
__global__ void packT_kernel(const float* __restrict__ src, h16* __restrict__ dst,
                             int K, int N, int Kp)
{
    long idx = (long)blockIdx.x * 256 + threadIdx.x;
    if (idx >= (long)N * Kp) return;
    int n = (int)(idx / Kp), k = (int)(idx % Kp);
    dst[idx] = (k < K) ? (h16)src[(size_t)k * N + n] : (h16)0.f;
}

// ---- concat [x | pos_emb | pad] -> f16 ------------------------------------
__global__ void cat_kernel(const float* __restrict__ x, const float* __restrict__ pe,
                           h16* __restrict__ out)
{
    long idx = (long)blockIdx.x * 256 + threadIdx.x;
    int row = (int)(idx / D1P_), j = (int)(idx % D1P_);
    int s = row % S_;
    float v = (j < H_) ? x[(size_t)row * H_ + j]
                       : (j < H_ + P_ ? pe[(size_t)s * P_ + (j - H_)] : 0.f);
    out[idx] = (h16)v;
}

// ---- LSTM recurrence: 16 batch rows per block, WMMA for h@Whh -------------
__launch_bounds__(256)
__global__ void lstm_kernel(const float* __restrict__ xg, const h16* __restrict__ whhT,
                            h16* __restrict__ hout, int diroff, int reverse)
{
    extern __shared__ char smem[];
    h16*   hs = (h16*)smem;                 // 16 x 264 f16 hidden
    float* cs = (float*)(hs + 16 * 264);    // 16 x 256 f32 cell
    float* gs = cs + 16 * 256;              // 16 x 1032 f32 gates
    int tid = threadIdx.x, wid = tid >> 5, lane = tid & 31;
    int b0 = blockIdx.x * 16;

    for (int i = tid; i < 16 * 264; i += 256) hs[i] = (h16)0.f;
    for (int i = tid; i < 16 * 256; i += 256) cs[i] = 0.f;
    __syncthreads();

    for (int tt = 0; tt < S_; ++tt) {
        int t = reverse ? (S_ - 1 - tt) : tt;
        v8f acc[8] = {};
#pragma unroll
        for (int kc = 0; kc < 8; ++kc) {
            v16h a = frag_ld(hs + kc * 32, 264);
#pragma unroll
            for (int nt = 0; nt < 8; ++nt) {
                v16h bf = frag_ld(whhT + (size_t)(wid * 128 + nt * 16) * LH_ + kc * 32, LH_);
                acc[nt] = wmma16(a, bf, acc[nt]);
            }
        }
        int rb = (lane >> 4) << 3;
        int cc = lane & 15;
#pragma unroll
        for (int nt = 0; nt < 8; ++nt) {
            int col = wid * 128 + nt * 16 + cc;
#pragma unroll
            for (int rr = 0; rr < 8; ++rr) {
                int row = rb + rr;
                gs[row * 1032 + col] =
                    acc[nt][rr] + xg[((size_t)(b0 + row) * S_ + t) * G4_ + col];
            }
        }
        __syncthreads();
        for (int i = tid; i < 16 * LH_; i += 256) {
            int row = i >> 8, j = i & 255;
            float gi = gs[row * 1032 + j];
            float gf = gs[row * 1032 + 256 + j];
            float gg = gs[row * 1032 + 512 + j];
            float go = gs[row * 1032 + 768 + j];
            float c = sigm_f(gf) * cs[i] + sigm_f(gi) * tanhf(gg);
            float h = sigm_f(go) * tanhf(c);
            cs[i] = c;
            hs[row * 264 + j] = (h16)h;
            hout[((size_t)(b0 + row) * S_ + t) * (2 * LH_) + diroff + j] = (h16)h;
        }
        __syncthreads();
    }
}

// ---- classifier: logits = h2 @ Wc + bc (N=57) -----------------------------
__launch_bounds__(64)
__global__ void cls_kernel(const h16* __restrict__ h2, const float* __restrict__ Wc,
                           const float* __restrict__ bc, float* __restrict__ logits)
{
    int row = blockIdx.x, n = threadIdx.x;
    if (n >= NL_) return;
    float s = bc[n];
    for (int k = 0; k < 2 * LH_; ++k)
        s += (float)h2[(size_t)row * (2 * LH_) + k] * Wc[(size_t)k * NL_ + n];
    logits[(size_t)row * NL_ + n] = s;
}

// ---- CRF NLL per batch element --------------------------------------------
__launch_bounds__(64)
__global__ void crf_kernel(const float* __restrict__ logits, const int* __restrict__ labels,
                           const int* __restrict__ mask, const float* __restrict__ trans,
                           const float* __restrict__ start_t, const float* __restrict__ end_t,
                           float* __restrict__ nll)
{
    __shared__ float alpha[NL_];
    __shared__ float num_s;
    int b = blockIdx.x, j = threadIdx.x;
    const float* lg = logits + (size_t)b * S_ * NL_;
    if (j < NL_) alpha[j] = start_t[j] + lg[j];
    if (j == 0) {
        const int* lab = labels + b * S_;
        const int* mk = mask + b * S_;
        float num = start_t[lab[0]] + lg[lab[0]];
        for (int t = 1; t < S_; ++t)
            if (mk[t]) num += trans[lab[t - 1] * NL_ + lab[t]] + lg[(size_t)t * NL_ + lab[t]];
        int cnt = 0;
        for (int t = 0; t < S_; ++t) cnt += (mk[t] ? 1 : 0);
        num += end_t[lab[cnt - 1]];
        num_s = num;
    }
    __syncthreads();
    for (int t = 1; t < S_; ++t) {
        float v = 0.f;
        if (j < NL_) {
            float mx = -1e30f;
            for (int i = 0; i < NL_; ++i) mx = fmaxf(mx, alpha[i] + trans[i * NL_ + j]);
            float s = 0.f;
            for (int i = 0; i < NL_; ++i) s += expf(alpha[i] + trans[i * NL_ + j] - mx);
            v = mx + logf(s) + lg[(size_t)t * NL_ + j];
        }
        __syncthreads();
        if (j < NL_ && mask[b * S_ + t]) alpha[j] = v;
        __syncthreads();
    }
    if (j == 0) {
        float mx = -1e30f;
        for (int i = 0; i < NL_; ++i) mx = fmaxf(mx, alpha[i] + end_t[i]);
        float s = 0.f;
        for (int i = 0; i < NL_; ++i) s += expf(alpha[i] + end_t[i] - mx);
        nll[b] = (mx + logf(s)) - num_s;
    }
}

__global__ void reduce_kernel(const float* __restrict__ nll, float* __restrict__ out)
{
    __shared__ float s[256];
    int tid = threadIdx.x;
    s[tid] = nll[tid];
    __syncthreads();
    for (int k = 128; k > 0; k >>= 1) { if (tid < k) s[tid] += s[tid + k]; __syncthreads(); }
    if (tid == 0) out[0] = s[0] / 256.0f;
}

// ===========================================================================
extern "C" void kernel_launch(void* const* d_in, const int* in_sizes, int n_in,
                              void* d_out, int out_size, void* d_ws, size_t ws_size,
                              hipStream_t stream)
{
    (void)in_sizes; (void)n_in; (void)out_size; (void)ws_size;
    char* W = (char*)d_ws;
    size_t off = 0;
    auto carve = [&](size_t bytes) {
        size_t r = off; off += (bytes + 255) & ~(size_t)255; return r;
    };
    float* x_f32   = (float*)(W + carve((size_t)BS_ * H_ * 4));
    float* tmp_f32 = (float*)(W + carve((size_t)BS_ * H_ * 4));
    h16*   x_f16   = (h16*)  (W + carve((size_t)BS_ * H_ * 2));
    char*  bufA    =          W + carve((size_t)BS_ * 2304 * 2);  // qkv f16 / xg_a f32
    char*  bufB    =          W + carve((size_t)BS_ * F_ * 2);    // ffn f16 / xg_b f32
    h16*   ctx_f16 = (h16*)  (W + carve((size_t)BS_ * H_ * 2));
    h16*   cat_f16 = (h16*)  (W + carve((size_t)BS_ * D1P_ * 2));
    h16*   h1_f16  = (h16*)  (W + carve((size_t)BS_ * 512 * 2));
    h16*   h2_f16  = (h16*)  (W + carve((size_t)BS_ * 512 * 2));
    float* logits  = (float*)(W + carve((size_t)BS_ * NL_ * 4));
    float* nll     = (float*)(W + carve(1024));
    // all weights stored transposed (N x K, f16) for scatter-free tile loads
    h16* wqkv16 = (h16*)(W + carve((size_t)L_ * H_ * 3 * H_ * 2));
    h16* wo16   = (h16*)(W + carve((size_t)L_ * H_ * H_ * 2));
    h16* w116   = (h16*)(W + carve((size_t)L_ * H_ * F_ * 2));
    h16* w216   = (h16*)(W + carve((size_t)L_ * F_ * H_ * 2));
    h16* wih1f  = (h16*)(W + carve((size_t)D1P_ * G4_ * 2));
    h16* wih1b  = (h16*)(W + carve((size_t)D1P_ * G4_ * 2));
    h16* wih2f  = (h16*)(W + carve((size_t)512 * G4_ * 2));
    h16* wih2b  = (h16*)(W + carve((size_t)512 * G4_ * 2));
    h16* whh1fT = (h16*)(W + carve((size_t)G4_ * LH_ * 2));
    h16* whh1bT = (h16*)(W + carve((size_t)G4_ * LH_ * 2));
    h16* whh2fT = (h16*)(W + carve((size_t)G4_ * LH_ * 2));
    h16* whh2bT = (h16*)(W + carve((size_t)G4_ * LH_ * 2));

    const int* ids   = (const int*)d_in[0];
    const int* amask = (const int*)d_in[1];
    const int* labs  = (const int*)d_in[2];

    auto PACKT = [&](const void* src, h16* dst, int K, int N, int Kp) {
        long total = (long)N * Kp;
        packT_kernel<<<dim3((unsigned)((total + 255) / 256)), dim3(256), 0, stream>>>(
            (const float*)src, dst, K, N, Kp);
    };
    for (int l = 0; l < L_; ++l) {
        PACKT((const float*)d_in[7]  + (size_t)l * H_ * 3 * H_, wqkv16 + (size_t)l * 3 * H_ * H_, H_, 3 * H_, H_);
        PACKT((const float*)d_in[9]  + (size_t)l * H_ * H_,     wo16   + (size_t)l * H_ * H_,     H_, H_,     H_);
        PACKT((const float*)d_in[13] + (size_t)l * H_ * F_,     w116   + (size_t)l * F_ * H_,     H_, F_,     H_);
        PACKT((const float*)d_in[15] + (size_t)l * F_ * H_,     w216   + (size_t)l * H_ * F_,     F_, H_,     F_);
    }
    PACKT(d_in[20], wih1f, H_ + P_, G4_, D1P_);
    PACKT(d_in[23], wih1b, H_ + P_, G4_, D1P_);
    PACKT(d_in[26], wih2f, 512,     G4_, 512);
    PACKT(d_in[29], wih2b, 512,     G4_, 512);
    PACKT(d_in[21], whh1fT, LH_, G4_, LH_);
    PACKT(d_in[24], whh1bT, LH_, G4_, LH_);
    PACKT(d_in[27], whh2fT, LH_, G4_, LH_);
    PACKT(d_in[30], whh2bT, LH_, G4_, LH_);

    // embeddings + LN0
    embed_kernel<<<BS_, 256, 0, stream>>>(ids, (const float*)d_in[3], (const float*)d_in[4], tmp_f32);
    ln_kernel<<<BS_, 256, 0, stream>>>(tmp_f32, nullptr, (const float*)d_in[5],
                                       (const float*)d_in[6], x_f32, x_f16);

    const size_t ATTN_SMEM = 128*72*2 + 128*72*2 + 64*136*2 + 128*132*4 + 128*136*2 + 128*4;
    const size_t LSTM_SMEM = 16*264*2 + 16*256*4 + 16*1032*4;

    // BERT layers
    for (int l = 0; l < L_; ++l) {
        gemm_kernel<<<dim3(3 * H_ / 64, BS_ / 64), 256, 0, stream>>>(
            x_f16, wqkv16 + (size_t)l * 3 * H_ * H_, (const float*)d_in[8] + (size_t)l * 3 * H_,
            nullptr, (h16*)bufA, BS_, 3 * H_, H_, 0);
        attn_kernel<<<B_ * NH_, 256, ATTN_SMEM, stream>>>((const h16*)bufA, amask, ctx_f16);
        gemm_kernel<<<dim3(H_ / 64, BS_ / 64), 256, 0, stream>>>(
            ctx_f16, wo16 + (size_t)l * H_ * H_, (const float*)d_in[10] + (size_t)l * H_,
            tmp_f32, nullptr, BS_, H_, H_, 0);
        ln_kernel<<<BS_, 256, 0, stream>>>(tmp_f32, x_f32,
            (const float*)d_in[11] + (size_t)l * H_, (const float*)d_in[12] + (size_t)l * H_,
            x_f32, x_f16);
        gemm_kernel<<<dim3(F_ / 64, BS_ / 64), 256, 0, stream>>>(
            x_f16, w116 + (size_t)l * F_ * H_, (const float*)d_in[14] + (size_t)l * F_,
            nullptr, (h16*)bufB, BS_, F_, H_, 1);
        gemm_kernel<<<dim3(H_ / 64, BS_ / 64), 256, 0, stream>>>(
            (const h16*)bufB, w216 + (size_t)l * H_ * F_, (const float*)d_in[16] + (size_t)l * H_,
            tmp_f32, nullptr, BS_, H_, F_, 0);
        ln_kernel<<<BS_, 256, 0, stream>>>(tmp_f32, x_f32,
            (const float*)d_in[17] + (size_t)l * H_, (const float*)d_in[18] + (size_t)l * H_,
            x_f32, x_f16);
    }

    // BiLSTM layer 1
    cat_kernel<<<(unsigned)((size_t)BS_ * D1P_ / 256), 256, 0, stream>>>(
        x_f32, (const float*)d_in[19], cat_f16);
    float* xg_a = (float*)bufA;
    float* xg_b = (float*)bufB;
    gemm_kernel<<<dim3(G4_ / 64, BS_ / 64), 256, 0, stream>>>(
        cat_f16, wih1f, (const float*)d_in[22], xg_a, nullptr, BS_, G4_, D1P_, 0);
    gemm_kernel<<<dim3(G4_ / 64, BS_ / 64), 256, 0, stream>>>(
        cat_f16, wih1b, (const float*)d_in[25], xg_b, nullptr, BS_, G4_, D1P_, 0);
    lstm_kernel<<<B_ / 16, 256, LSTM_SMEM, stream>>>(xg_a, whh1fT, h1_f16, 0, 0);
    lstm_kernel<<<B_ / 16, 256, LSTM_SMEM, stream>>>(xg_b, whh1bT, h1_f16, LH_, 1);

    // BiLSTM layer 2
    gemm_kernel<<<dim3(G4_ / 64, BS_ / 64), 256, 0, stream>>>(
        h1_f16, wih2f, (const float*)d_in[28], xg_a, nullptr, BS_, G4_, 512, 0);
    gemm_kernel<<<dim3(G4_ / 64, BS_ / 64), 256, 0, stream>>>(
        h1_f16, wih2b, (const float*)d_in[31], xg_b, nullptr, BS_, G4_, 512, 0);
    lstm_kernel<<<B_ / 16, 256, LSTM_SMEM, stream>>>(xg_a, whh2fT, h2_f16, 0, 0);
    lstm_kernel<<<B_ / 16, 256, LSTM_SMEM, stream>>>(xg_b, whh2bT, h2_f16, LH_, 1);

    // classifier + CRF + mean
    cls_kernel<<<BS_, 64, 0, stream>>>(h2_f16, (const float*)d_in[32], (const float*)d_in[33], logits);
    crf_kernel<<<B_, 64, 0, stream>>>(logits, labs, amask, (const float*)d_in[34],
                                      (const float*)d_in[35], (const float*)d_in[36], nll);
    reduce_kernel<<<1, 256, 0, stream>>>(nll, (float*)d_out);
}
